// GC_withres_9852654977352
// MI455X (gfx1250) — matrix operand order repaired and checked by reading, not compiled, not run on hardware
//
#include <hip/hip_runtime.h>

#define D 128
#define SMOOTHF 0.5f

typedef __attribute__((ext_vector_type(16))) __bf16 v16bf;
typedef __attribute__((ext_vector_type(8)))  __bf16 v8bf;
typedef __attribute__((ext_vector_type(8)))  float  v8f;

// round-to-nearest-even float -> bf16 bits
static __device__ __forceinline__ unsigned bfbits(float f) {
    unsigned u = __builtin_bit_cast(unsigned, f);
    u += 0x7fffu + ((u >> 16) & 1u);
    return u >> 16;
}

// pack two consecutive f32 into one dword of two bf16 (bandwidth-bound, one-shot)
__global__ void cvt_bf16_kernel(const float* __restrict__ src,
                                unsigned* __restrict__ dst, long long npairs) {
    long long i = (long long)blockIdx.x * blockDim.x + threadIdx.x;
    if (i >= npairs) return;
    float2 v = ((const float2*)src)[i];
    dst[i] = (bfbits(v.y) << 16) | bfbits(v.x);
}

__global__ void init_deg_kernel(float* __restrict__ deg, int N) {
    int i = blockIdx.x * blockDim.x + threadIdx.x;
    if (i < N) deg[i] = 1.0f;   // identity term of A_gcn = adj + I
}

__global__ void accum_deg_kernel(const int* __restrict__ ecol,
                                 const float* __restrict__ eval,
                                 float* __restrict__ deg, int E) {
    int e = blockIdx.x * blockDim.x + threadIdx.x;
    if (e < E) atomicAdd(&deg[ecol[e]], eval[e]);
}

__global__ void rsqrt_deg_kernel(float* __restrict__ deg, int N) {
    int i = blockIdx.x * blockDim.x + threadIdx.x;
    if (i < N) deg[i] = rsqrtf(deg[i]);   // d = deg^{-1/2} in place
}

// support = x @ W^T + b  via v_wmma_f32_16x16x32_bf16 on preconverted operands.
// Also seeds out = support * dinv  (the "+ I@h" term of agg).
__global__ void __launch_bounds__(256)
gemm_support_kernel(const __bf16* __restrict__ xb, const __bf16* __restrict__ wb,
                    const float* __restrict__ bias, const float* __restrict__ dinv,
                    float* __restrict__ support, float* __restrict__ out,
                    int N, int rowTiles) {
    int wave = threadIdx.x >> 5;
    int lane = threadIdx.x & 31;
    int tile = blockIdx.x * 8 + wave;               // 8 waves / block
    if (tile >= rowTiles * 8) return;               // wave-uniform
    int n0 = (tile >> 3) << 4;                      // row tile base
    int o0 = (tile & 7) << 4;                       // col tile base (D_OUT/16 = 8)
    int hf = lane >> 4;                             // 16-lane half
    int lm = lane & 15;

    int m     = n0 + lm;
    int msafe = (m < N) ? m : 0;                    // clamp loads; EXEC stays all-1
    const __bf16* arow = xb + (size_t)msafe * D;
    const __bf16* brow = wb + (size_t)(o0 + lm) * D;  // W row-major: contiguous in K

    v8f c = {};
#pragma unroll
    for (int kt = 0; kt < 4; ++kt) {
        int k0 = kt * 32;
        // A 16x32 bf16: lane=M; elems 0..7 = K in [k0+hf*8, +8), elems 8..15 = +16
        const __bf16* ap = arow + k0 + hf * 8;
        v8bf alo = *(const v8bf*)ap;                // 16 B load
        v8bf ahi = *(const v8bf*)(ap + 16);         // 16 B load
        v16bf a = __builtin_shufflevector(alo, ahi,
                      0, 1, 2, 3, 4, 5, 6, 7, 8, 9, 10, 11, 12, 13, 14, 15);
        // B 32x16 bf16: lane=N col; elems 0..7 = K in [k0+hf*8, +8), elems 8..15 = +16
        const __bf16* bp = brow + k0 + hf * 8;
        v8bf blo = *(const v8bf*)bp;                // 16 B load
        v8bf bhi = *(const v8bf*)(bp + 16);         // 16 B load
        v16bf b = __builtin_shufflevector(blo, bhi,
                      0, 1, 2, 3, 4, 5, 6, 7, 8, 9, 10, 11, 12, 13, 14, 15);

        c = __builtin_amdgcn_wmma_f32_16x16x32_bf16(
                false, a, false, b, (short)0, c, false, false);
    }

    // Epilogue. C layout: VGPR r -> row n0 + r + 8*hf, col = o0 + lm.
    float bv = bias[o0 + lm];
    int rbase = n0 + hf * 8;
    size_t idx0 = (size_t)rbase * D + o0 + lm;

    if (n0 + 16 <= N) {                             // wave-uniform full tile
        const float4* dv = (const float4*)(dinv + rbase);   // 32B-aligned
        float4 d0 = dv[0], d1 = dv[1];
        float dr[8] = {d0.x, d0.y, d0.z, d0.w, d1.x, d1.y, d1.z, d1.w};
#pragma unroll
        for (int r = 0; r < 8; ++r) {
            float s = c[r] + bv;
            size_t idx = idx0 + (size_t)r * D;
            support[idx] = s;
            out[idx]     = s * dr[r];               // identity/h term of agg
        }
    } else {                                        // partial last tile (rare)
#pragma unroll
        for (int r = 0; r < 8; ++r) {
            int row = rbase + r;
            if (row < N) {
                float s = c[r] + bv;
                size_t idx = idx0 + (size_t)r * D;
                support[idx] = s;
                out[idx]     = s * dinv[row];
            }
        }
    }
}

// out[row] += edge_val * dinv[col] * support[col]   (1 wave per edge; L2-resident)
__global__ void __launch_bounds__(256)
scatter_kernel(const int* __restrict__ erow, const int* __restrict__ ecol,
               const float* __restrict__ eval, const float* __restrict__ dinv,
               const float* __restrict__ support, float* __restrict__ out, int E) {
    int gid  = blockIdx.x * blockDim.x + threadIdx.x;
    int e    = gid >> 5;
    int lane = gid & 31;
    if (e >= E) return;
    int col = ecol[e];
    int row = erow[e];
    float s = eval[e] * dinv[col];                  // fold right-scale into weight
    const float4* src = (const float4*)(support + (size_t)col * D);
    float4 v = src[lane];                           // 32 lanes x float4 = 128 feats
    float* dst = out + (size_t)row * D + lane * 4;
    atomicAdd(dst + 0, v.x * s);
    atomicAdd(dst + 1, v.y * s);
    atomicAdd(dst + 2, v.z * s);
    atomicAdd(dst + 3, v.w * s);
}

// out = (out * dinv[n] * S + support) / (1 + S)
__global__ void finalize_kernel(const float* __restrict__ support,
                                const float* __restrict__ dinv,
                                float* __restrict__ out, long long total) {
    long long i = (long long)blockIdx.x * blockDim.x + threadIdx.x;
    if (i >= total) return;
    int n = (int)(i >> 7);                          // D = 128
    out[i] = (out[i] * dinv[n] * SMOOTHF + support[i]) * (1.0f / (1.0f + SMOOTHF));
}

static inline size_t align256(size_t v) { return (v + 255) & ~(size_t)255; }

extern "C" void kernel_launch(void* const* d_in, const int* in_sizes, int n_in,
                              void* d_out, int out_size, void* d_ws, size_t ws_size,
                              hipStream_t stream) {
    const float* x    = (const float*)d_in[0];
    const int*   erow = (const int*)d_in[1];
    const int*   ecol = (const int*)d_in[2];
    const float* eval = (const float*)d_in[3];
    const float* W    = (const float*)d_in[4];
    const float* bias = (const float*)d_in[5];
    float*       out  = (float*)d_out;

    int N = in_sizes[0] / D;
    int E = in_sizes[1];

    // workspace: dinv[N] f32 | support[N*D] f32 | xb[N*D] bf16 | Wb[D*D] bf16
    char* ws = (char*)d_ws;
    size_t o_dinv = 0;
    size_t o_sup  = align256(o_dinv + (size_t)N * 4);
    size_t o_xb   = align256(o_sup + (size_t)N * D * 4);
    size_t o_wb   = align256(o_xb + (size_t)N * D * 2);
    float*  dinv    = (float*)(ws + o_dinv);
    float*  support = (float*)(ws + o_sup);
    __bf16* xb      = (__bf16*)(ws + o_xb);
    __bf16* wb      = (__bf16*)(ws + o_wb);

    // one-shot operand conversion (bandwidth-bound)
    long long xpairs = (long long)N * D / 2;
    cvt_bf16_kernel<<<(unsigned)((xpairs + 255) / 256), 256, 0, stream>>>(
        x, (unsigned*)xb, xpairs);
    long long wpairs = (long long)D * D / 2;
    cvt_bf16_kernel<<<(unsigned)((wpairs + 255) / 256), 256, 0, stream>>>(
        W, (unsigned*)wb, wpairs);

    init_deg_kernel<<<(N + 255) / 256, 256, 0, stream>>>(dinv, N);
    accum_deg_kernel<<<(E + 255) / 256, 256, 0, stream>>>(ecol, eval, dinv, E);
    rsqrt_deg_kernel<<<(N + 255) / 256, 256, 0, stream>>>(dinv, N);

    int rowTiles = (N + 15) / 16;                   // 3125 for N=50000
    gemm_support_kernel<<<rowTiles, 256, 0, stream>>>(xb, wb, bias, dinv,
                                                      support, out, N, rowTiles);

    long long sthreads = (long long)E * 32;
    scatter_kernel<<<(unsigned)((sthreads + 255) / 256), 256, 0, stream>>>(
        erow, ecol, eval, dinv, support, out, E);

    long long total = (long long)N * D;
    finalize_kernel<<<(unsigned)((total + 255) / 256), 256, 0, stream>>>(
        support, dinv, out, total);
}
